// PISNNModel_13134009991682
// MI455X (gfx1250) — matrix operand order repaired and checked by reading, not compiled, not run on hardware
//
#include <hip/hip_runtime.h>

// ---------------------------------------------------------------------------
// Types / helpers
// ---------------------------------------------------------------------------
typedef __attribute__((ext_vector_type(16))) __bf16 v16bf;
typedef __attribute__((ext_vector_type(8)))  __bf16 v8bf;
typedef __attribute__((ext_vector_type(8)))  float  v8f;

__device__ __forceinline__ __bf16 f2bf(float f) {
    union { float f; unsigned u; } x; x.f = f;
    unsigned r = (x.u + 0x7fffu + ((x.u >> 16) & 1u)) >> 16;
    unsigned short s = (unsigned short)r;
    return __builtin_bit_cast(__bf16, s);
}
__device__ __forceinline__ float bf2f(__bf16 b) {
    unsigned short s = __builtin_bit_cast(unsigned short, b);
    union { unsigned u; float f; } x; x.u = ((unsigned)s) << 16;
    return x.f;
}
__device__ __forceinline__ __bf16 bf_zero() {
    return __builtin_bit_cast(__bf16, (unsigned short)0);
}

static inline long cdivl(long a, long b) { return (a + b - 1) / b; }

// ---------------------------------------------------------------------------
// WMMA GEMM: Y[M x (NT*16)] = act( X[M x Kpad](bf16) * Wpanel + bias )
// X row-major bf16, Kpad multiple of 32.
// Wpanel layout: [(ks*NT + nt)*32 + lane]*16 + e  where
//   B element (K = ks*32 + (lane<16?0:16) + e, N = nt*16 + (lane&15))
// One wave (32 lanes) computes a 16-row x NT*16-col tile.
// ---------------------------------------------------------------------------
template <int NT, int ACT>
__global__ __launch_bounds__(128) void gemm_bf16_wmma(
    const __bf16* __restrict__ X, const __bf16* __restrict__ Wp,
    const float* __restrict__ bias, int bias_n,
    float* __restrict__ outF, int ldF, int vcF,
    __bf16* __restrict__ outB, int ldB, int vcB,
    int M, int Kpad, int Ksteps)
{
    const int wave = threadIdx.x >> 5;
    const int lane = threadIdx.x & 31;
    const long rb   = (long)blockIdx.x * 4 + wave;   // 16-row tile index
    const long row0 = rb * 16;
    if (row0 >= M) return;                            // wave-uniform exit

    const int  hs   = lane >> 4;                      // half-select 0/1
    const int  rlo  = lane & 15;
    long arow = row0 + rlo;
    if (arow >= M) arow = (long)M - 1;                // clamp loads, guard stores
    const __bf16* xrow = X + arow * (long)Kpad;

    v8f acc[NT];
#pragma unroll
    for (int t = 0; t < NT; ++t)
#pragma unroll
        for (int i = 0; i < 8; ++i) acc[t][i] = 0.0f;

    for (int ks = 0; ks < Ksteps; ++ks) {
        // A tile 16x32 bf16: lane (row = rlo) holds K = kb..kb+15 (kb = hs*16)
        v8bf a0 = *(const v8bf*)(xrow + ks * 32 + hs * 16);
        v8bf a1 = *(const v8bf*)(xrow + ks * 32 + hs * 16 + 8);
        v16bf a;
#pragma unroll
        for (int i = 0; i < 8; ++i) { a[i] = a0[i]; a[8 + i] = a1[i]; }

        const __bf16* wb = Wp + ((long)ks * NT) * 512 + lane * 16;
#pragma unroll
        for (int t = 0; t < NT; ++t) {
            v16bf b = *(const v16bf*)(wb + (long)t * 512);
            acc[t] = __builtin_amdgcn_wmma_f32_16x16x32_bf16(
                false, a, false, b, (short)0, acc[t], false, false);
        }
    }

    // C layout: lanes 0-15: (M = r, N = lane); lanes 16-31: (M = r+8, N = lane-16)
#pragma unroll
    for (int t = 0; t < NT; ++t) {
        const int col = t * 16 + rlo;
        const float bv = (col < bias_n) ? bias[col] : 0.0f;
#pragma unroll
        for (int r = 0; r < 8; ++r) {
            const long row = row0 + r + hs * 8;
            if (row >= M) continue;
            float v = acc[t][r] + bv;
            if (ACT) v = tanhf(v);
            if (outF && col < vcF) outF[row * (long)ldF + col] = v;
            if (outB && col < vcB) outB[row * (long)ldB + col] = f2bf(v);
        }
    }
}

// Pack f32 weights (Kreal x Ncols, row-major) into the WMMA B-panel (bf16).
__global__ void pack_w_kernel(const float* __restrict__ W, __bf16* __restrict__ P,
                              int Ksteps, int NT, int Kreal, int Ncols)
{
    int idx = blockIdx.x * blockDim.x + threadIdx.x;
    int total = Ksteps * NT * 512;
    if (idx >= total) return;
    int e    = idx & 15;
    int lane = (idx >> 4) & 31;
    int blk  = idx >> 9;            // ks*NT + nt
    int nt   = blk % NT;
    int ks   = blk / NT;
    int k = ks * 32 + (lane >> 4) * 16 + e;
    int n = nt * 16 + (lane & 15);
    float v = (k < Kreal && n < Ncols) ? W[(long)k * Ncols + n] : 0.0f;
    P[idx] = f2bf(v);
}

// ---------------------------------------------------------------------------
// Elementwise / scatter kernels
// ---------------------------------------------------------------------------
__global__ void count_scatter_kernel(const int* __restrict__ idxs,
                                     float* __restrict__ deg, long n)
{
    long i = (long)blockIdx.x * blockDim.x + threadIdx.x;
    if (i < n) atomicAdd(deg + idxs[i], 1.0f);
}

__global__ void clamp1_kernel(float* __restrict__ p, long n)
{
    long i = (long)blockIdx.x * blockDim.x + threadIdx.x;
    if (i < n) p[i] = fmaxf(p[i], 1.0f);
}

// encoder stage 1: h = tanh(node_in @ enc_W1 + b1)  (K=4, VALU), output bf16
__global__ void enc1_kernel(const float* __restrict__ uv, const float* __restrict__ curv,
                            const float* __restrict__ W1, const float* __restrict__ b1,
                            __bf16* __restrict__ h, long N)
{
    long i = (long)blockIdx.x * blockDim.x + threadIdx.x;
    if (i >= N * 64) return;
    long n = i >> 6; int c = (int)(i & 63);
    float x0 = uv[n * 2], x1 = uv[n * 2 + 1];
    float x2 = curv[n * 2], x3 = curv[n * 2 + 1];
    float v = b1[c] + x0 * W1[c] + x1 * W1[64 + c] + x2 * W1[128 + c] + x3 * W1[192 + c];
    h[i] = f2bf(tanhf(v));
}

// SpMM scatter, 64-wide features: thread = (nnz, 16-col chunk)
__global__ void spmm64_kernel(const int* __restrict__ rows, const int* __restrict__ cols,
                              const float* __restrict__ vals, const float* __restrict__ x,
                              float* __restrict__ y, long nnz)
{
    long i = (long)blockIdx.x * blockDim.x + threadIdx.x;
    if (i >= nnz * 4) return;
    long n = i >> 2; int ch = (int)(i & 3);
    int r = rows[n], c = cols[n];
    float v = vals[n];
    const float* xr = x + (long)c * 64 + ch * 16;
    float* yr = y + (long)r * 64 + ch * 16;
#pragma unroll
    for (int j = 0; j < 16; ++j) atomicAdd(yr + j, v * xr[j]);
}

// SpMM scatter, 4-wide features: thread = nnz
__global__ void spmm4_kernel(const int* __restrict__ rows, const int* __restrict__ cols,
                             const float* __restrict__ vals, const float* __restrict__ x,
                             float* __restrict__ y, long nnz)
{
    long n = (long)blockIdx.x * blockDim.x + threadIdx.x;
    if (n >= nnz) return;
    int r = rows[n], c = cols[n];
    float v = vals[n];
#pragma unroll
    for (int j = 0; j < 4; ++j)
        atomicAdd(y + (long)r * 4 + j, v * x[(long)c * 4 + j]);
}

// edge->node messages: msg = ef @ e2n_W + b (K=4, fused), scatter to src & dst
__global__ void e2n_scatter_kernel(const float* __restrict__ ef, const int* __restrict__ edges,
                                   const float* __restrict__ W, const float* __restrict__ B,
                                   float* __restrict__ nfe, long E)
{
    long i = (long)blockIdx.x * blockDim.x + threadIdx.x;
    if (i >= E * 4) return;
    long e = i >> 2; int ch = (int)(i & 3);
    float f0 = ef[e * 4], f1 = ef[e * 4 + 1], f2 = ef[e * 4 + 2], f3 = ef[e * 4 + 3];
    int s = edges[e * 2], d = edges[e * 2 + 1];
#pragma unroll
    for (int j = 0; j < 16; ++j) {
        int c = ch * 16 + j;
        float m = B[c] + f0 * W[c] + f1 * W[64 + c] + f2 * W[128 + c] + f3 * W[192 + c];
        atomicAdd(nfe + (long)s * 64 + c, m);
        atomicAdd(nfe + (long)d * 64 + c, m);
    }
}

// tri->edge messages: msg = tf @ t2e_W + b (K=4, fused), scatter to 3 edges
__global__ void t2e_scatter_kernel(const float* __restrict__ tf, const int* __restrict__ te,
                                   const float* __restrict__ W, const float* __restrict__ B,
                                   float* __restrict__ acc, long T)
{
    long i = (long)blockIdx.x * blockDim.x + threadIdx.x;
    if (i >= T * 4) return;
    long t = i >> 2; int ch = (int)(i & 3);
    float f0 = tf[t * 4], f1 = tf[t * 4 + 1], f2 = tf[t * 4 + 2], f3 = tf[t * 4 + 3];
    int e0 = te[t * 3], e1 = te[t * 3 + 1], e2 = te[t * 3 + 2];
#pragma unroll
    for (int j = 0; j < 16; ++j) {
        int c = ch * 16 + j;
        float m = B[c] + f0 * W[c] + f1 * W[64 + c] + f2 * W[128 + c] + f3 * W[192 + c];
        atomicAdd(acc + (long)e0 * 64 + c, m);
        atomicAdd(acc + (long)e1 * 64 + c, m);
        atomicAdd(acc + (long)e2 * 64 + c, m);
    }
}

// Build Xe (E x 96 bf16): [ef(4) | acc/tri_deg + 0.5*(nm[src]+nm[dst]) (64) | 0 pad(28)]
__global__ void edge_combine_kernel(const float* __restrict__ acc, const float* __restrict__ tdeg,
                                    const float* __restrict__ nm, const int* __restrict__ edges,
                                    const float* __restrict__ ef, __bf16* __restrict__ Xe, long E)
{
    long i = (long)blockIdx.x * blockDim.x + threadIdx.x;
    if (i >= E * 4) return;
    long e = i >> 2; int ch = (int)(i & 3);
    int s = edges[e * 2], d = edges[e * 2 + 1];
    float td = tdeg[e];
    __bf16* xr = Xe + e * 96;
#pragma unroll
    for (int j = 0; j < 16; ++j) {
        int c = ch * 16 + j;
        float v = acc[e * 64 + c] / td
                + 0.5f * (nm[(long)s * 64 + c] + nm[(long)d * 64 + c]);
        xr[4 + c] = f2bf(v);
    }
    if (ch == 0) {
#pragma unroll
        for (int k = 0; k < 4; ++k) xr[k] = f2bf(ef[e * 4 + k]);
    }
    if (ch == 3) {
#pragma unroll
        for (int k = 68; k < 96; ++k) xr[k] = bf_zero();
    }
}

// Build Xt (T x 96 bf16): [tf(4) | mean(ef over 3 edges)@e2t_W + b (64) | 0 pad]
__global__ void tri_build_kernel(const float* __restrict__ ef, const float* __restrict__ tf,
                                 const int* __restrict__ te, const float* __restrict__ W,
                                 const float* __restrict__ B, __bf16* __restrict__ Xt, long T)
{
    long i = (long)blockIdx.x * blockDim.x + threadIdx.x;
    if (i >= T * 4) return;
    long t = i >> 2; int ch = (int)(i & 3);
    int e0 = te[t * 3], e1 = te[t * 3 + 1], e2 = te[t * 3 + 2];
    const float k3 = 1.0f / 3.0f;
    float a0 = (ef[(long)e0 * 4 + 0] + ef[(long)e1 * 4 + 0] + ef[(long)e2 * 4 + 0]) * k3;
    float a1 = (ef[(long)e0 * 4 + 1] + ef[(long)e1 * 4 + 1] + ef[(long)e2 * 4 + 1]) * k3;
    float a2 = (ef[(long)e0 * 4 + 2] + ef[(long)e1 * 4 + 2] + ef[(long)e2 * 4 + 2]) * k3;
    float a3 = (ef[(long)e0 * 4 + 3] + ef[(long)e1 * 4 + 3] + ef[(long)e2 * 4 + 3]) * k3;
    __bf16* xr = Xt + t * 96;
#pragma unroll
    for (int j = 0; j < 16; ++j) {
        int c = ch * 16 + j;
        float m = B[c] + a0 * W[c] + a1 * W[64 + c] + a2 * W[128 + c] + a3 * W[192 + c];
        xr[4 + c] = f2bf(m);
    }
    if (ch == 0) {
#pragma unroll
        for (int k = 0; k < 4; ++k) xr[k] = f2bf(tf[t * 4 + k]);
    }
    if (ch == 3) {
#pragma unroll
        for (int k = 68; k < 96; ++k) xr[k] = bf_zero();
    }
}

// Build Xn (N x 128 bf16): [nf (64) | nfe/deg (64)]
__global__ void xn_build_kernel(const __bf16* __restrict__ nfb, const float* __restrict__ nfe,
                                const float* __restrict__ deg, __bf16* __restrict__ Xn, long N)
{
    long i = (long)blockIdx.x * blockDim.x + threadIdx.x;
    if (i >= N * 64) return;
    long n = i >> 6; int c = (int)(i & 63);
    Xn[n * 128 + c]      = nfb[i];
    Xn[n * 128 + 64 + c] = f2bf(nfe[i] / deg[n]);
}

__global__ void node_update_kernel(float* __restrict__ nf, __bf16* __restrict__ nfb,
                                   const float* __restrict__ upd, const float* __restrict__ sm,
                                   long n64)
{
    long i = (long)blockIdx.x * blockDim.x + threadIdx.x;
    if (i >= n64) return;
    float v = nf[i] + 0.1f * upd[i] - 0.05f * sm[i];
    nf[i]  = v;
    nfb[i] = f2bf(v);
}

// 4-wide feature residual update; MLP output has row stride 16 (WMMA tile)
__global__ void feat4_update_kernel(float* __restrict__ f, const float* __restrict__ upd,
                                    const float* __restrict__ sm, long M)
{
    long i = (long)blockIdx.x * blockDim.x + threadIdx.x;
    if (i >= M * 4) return;
    long e = i >> 2; int j = (int)(i & 3);
    f[i] = f[i] + 0.1f * upd[e * 16 + j] - 0.05f * sm[i];
}

// decoder head (64->2) + FitzHugh-Nagumo RHS
__global__ void final_kernel(const __bf16* __restrict__ hd, const float* __restrict__ W2,
                             const float* __restrict__ b2, const float* __restrict__ uv,
                             float* __restrict__ out, long N)
{
    long n = (long)blockIdx.x * blockDim.x + threadIdx.x;
    if (n >= N) return;
    float a0 = b2[0], a1 = b2[1];
    const __bf16* hr = hd + n * 64;
#pragma unroll 4
    for (int c = 0; c < 64; ++c) {
        float h = bf2f(hr[c]);
        a0 += h * W2[c * 2 + 0];
        a1 += h * W2[c * 2 + 1];
    }
    float u = uv[n * 2], v = uv[n * 2 + 1];
    float du = (0.1f - u) * (u - 1.0f) * u - v + 1.0e-4f * a0;
    float dv = 0.01f * (0.5f * u - v) + 1.0e-5f * a1;
    out[n * 2 + 0] = du;
    out[n * 2 + 1] = dv;
}

// ---------------------------------------------------------------------------
// Host orchestration
// ---------------------------------------------------------------------------
extern "C" void kernel_launch(void* const* d_in, const int* in_sizes, int n_in,
                              void* d_out, int out_size, void* d_ws, size_t ws_size,
                              hipStream_t stream)
{
    (void)n_in; (void)out_size; (void)ws_size;

    const float* uv        = (const float*)d_in[0];
    const float* curv      = (const float*)d_in[1];
    const float* ef_in     = (const float*)d_in[2];
    const float* tf_in     = (const float*)d_in[3];
    const int*   edges     = (const int*)d_in[4];
    const int*   tri_edges = (const int*)d_in[5];
    const int*   l0r = (const int*)d_in[6];  const int* l0c = (const int*)d_in[7];
    const float* l0v = (const float*)d_in[8];
    const int*   l1r = (const int*)d_in[9];  const int* l1c = (const int*)d_in[10];
    const float* l1v = (const float*)d_in[11];
    const int*   l2r = (const int*)d_in[12]; const int* l2c = (const int*)d_in[13];
    const float* l2v = (const float*)d_in[14];
    // params in _make_params insertion order
    const float* enc_W1 = (const float*)d_in[15]; const float* enc_b1 = (const float*)d_in[16];
    const float* enc_W2 = (const float*)d_in[17]; const float* enc_b2 = (const float*)d_in[18];
    const float* dec_W1 = (const float*)d_in[19]; const float* dec_b1 = (const float*)d_in[20];
    const float* dec_W2 = (const float*)d_in[21]; const float* dec_b2 = (const float*)d_in[22];
    const float* nu_W1 = (const float*)d_in[23]; const float* nu_b1 = (const float*)d_in[24];
    const float* nu_W2 = (const float*)d_in[25]; const float* nu_b2 = (const float*)d_in[26];
    const float* eu_W1 = (const float*)d_in[27]; const float* eu_b1 = (const float*)d_in[28];
    const float* eu_W2 = (const float*)d_in[29]; const float* eu_b2 = (const float*)d_in[30];
    const float* tu_W1 = (const float*)d_in[31]; const float* tu_b1 = (const float*)d_in[32];
    const float* tu_W2 = (const float*)d_in[33]; const float* tu_b2 = (const float*)d_in[34];
    const float* e2n_W = (const float*)d_in[35]; const float* e2n_b = (const float*)d_in[36];
    const float* n2e_W = (const float*)d_in[37]; const float* n2e_b = (const float*)d_in[38];
    const float* t2e_W = (const float*)d_in[39]; const float* t2e_b = (const float*)d_in[40];
    const float* e2t_W = (const float*)d_in[41]; const float* e2t_b = (const float*)d_in[42];

    const long nN = in_sizes[0] / 2;
    const long nE = in_sizes[2] / 4;
    const long nT = in_sizes[3] / 4;
    const long NNZ0 = in_sizes[6], NNZ1 = in_sizes[9], NNZ2 = in_sizes[12];
    const int NL = 3;

    // --- workspace bump allocator ---
    char* wp = (char*)d_ws;
    auto alloc = [&](size_t bytes) -> char* {
        char* p = wp; wp += (bytes + 255) & ~(size_t)255; return p;
    };
    float*  nf   = (float*)alloc(nN * 64 * 4);
    __bf16* nfb  = (__bf16*)alloc(nN * 64 * 2);
    float*  ef   = (float*)alloc(nE * 4 * 4);
    float*  tf   = (float*)alloc(nT * 4 * 4);
    float*  deg  = (float*)alloc(nN * 4);
    float*  tdeg = (float*)alloc(nE * 4);
    __bf16* hn   = (__bf16*)alloc(nN * 64 * 2);   // enc hidden / node MLP hidden / dec hidden
    float*  ns   = (float*)alloc(nN * 64 * 4);
    float*  es   = (float*)alloc(nE * 4 * 4);
    float*  ts   = (float*)alloc(nT * 4 * 4);
    float*  nfe  = (float*)alloc(nN * 64 * 4);
    float*  nmsg = (float*)alloc(nN * 64 * 4);    // n2e output; reused as node-MLP output
    float*  eacc = (float*)alloc(nE * 64 * 4);    // tri->edge accum; reused as edge-MLP out (ld16)
    __bf16* Xn   = (__bf16*)alloc(nN * 128 * 2);
    __bf16* Xe   = (__bf16*)alloc(nE * 96 * 2);
    __bf16* he   = (__bf16*)alloc(nE * 64 * 2);
    __bf16* Xt   = (__bf16*)alloc(nT * 96 * 2);   // reused as tri-MLP out f32 (ld16)
    __bf16* ht   = (__bf16*)alloc(nT * 64 * 2);

    __bf16* Penc2 = (__bf16*)alloc(4096 * 2);
    __bf16* Pdec1 = (__bf16*)alloc(4096 * 2);
    __bf16 *Pn2e[3], *Pnu1[3], *Pnu2[3], *Peu1[3], *Peu2[3], *Ptu1[3], *Ptu2[3];
    for (int l = 0; l < NL; ++l) {
        Pn2e[l] = (__bf16*)alloc(4096 * 2);   // K=64  NT=4
        Pnu1[l] = (__bf16*)alloc(8192 * 2);   // K=128 NT=4
        Pnu2[l] = (__bf16*)alloc(4096 * 2);   // K=64  NT=4
        Peu1[l] = (__bf16*)alloc(6144 * 2);   // K=96  NT=4 (Kreal 68)
        Peu2[l] = (__bf16*)alloc(1024 * 2);   // K=64  NT=1 (Ncols 4)
        Ptu1[l] = (__bf16*)alloc(6144 * 2);
        Ptu2[l] = (__bf16*)alloc(1024 * 2);
    }

    const int BLK = 256;
    auto grid1 = [&](long total) { return dim3((unsigned)cdivl(total, BLK)); };

    auto pack = [&](const float* W, __bf16* P, int Kpad, int Kreal, int NT, int Ncols) {
        int total = (Kpad / 32) * NT * 512;
        pack_w_kernel<<<grid1(total), BLK, 0, stream>>>(W, P, Kpad / 32, NT, Kreal, Ncols);
    };
    auto gemm = [&](const __bf16* X, const __bf16* Wp, const float* bias, int bias_n,
                    float* outF, int ldF, int vcF, __bf16* outB, int ldB, int vcB,
                    long M, int Kpad, int NT, int act) {
        unsigned blocks = (unsigned)cdivl(cdivl(M, 16), 4);
        if (NT == 4 && act == 1)
            gemm_bf16_wmma<4, 1><<<blocks, 128, 0, stream>>>(X, Wp, bias, bias_n,
                outF, ldF, vcF, outB, ldB, vcB, (int)M, Kpad, Kpad / 32);
        else if (NT == 4)
            gemm_bf16_wmma<4, 0><<<blocks, 128, 0, stream>>>(X, Wp, bias, bias_n,
                outF, ldF, vcF, outB, ldB, vcB, (int)M, Kpad, Kpad / 32);
        else if (act == 1)
            gemm_bf16_wmma<1, 1><<<blocks, 128, 0, stream>>>(X, Wp, bias, bias_n,
                outF, ldF, vcF, outB, ldB, vcB, (int)M, Kpad, Kpad / 32);
        else
            gemm_bf16_wmma<1, 0><<<blocks, 128, 0, stream>>>(X, Wp, bias, bias_n,
                outF, ldF, vcF, outB, ldB, vcB, (int)M, Kpad, Kpad / 32);
    };

    // --- pack all weights (cheap, deterministic every call) ---
    pack(enc_W2, Penc2, 64, 64, 4, 64);
    pack(dec_W1, Pdec1, 64, 64, 4, 64);
    for (int l = 0; l < NL; ++l) {
        pack(n2e_W + (long)l * 64 * 64, Pn2e[l], 64, 64, 4, 64);
        pack(nu_W1 + (long)l * 128 * 64, Pnu1[l], 128, 128, 4, 64);
        pack(nu_W2 + (long)l * 64 * 64, Pnu2[l], 64, 64, 4, 64);
        pack(eu_W1 + (long)l * 68 * 64, Peu1[l], 96, 68, 4, 64);
        pack(eu_W2 + (long)l * 64 * 4, Peu2[l], 64, 64, 1, 4);
        pack(tu_W1 + (long)l * 68 * 64, Ptu1[l], 96, 68, 4, 64);
        pack(tu_W2 + (long)l * 64 * 4, Ptu2[l], 64, 64, 1, 4);
    }

    // --- degrees ---
    hipMemsetAsync(deg, 0, nN * 4, stream);
    hipMemsetAsync(tdeg, 0, nE * 4, stream);
    count_scatter_kernel<<<grid1(2 * nE), BLK, 0, stream>>>(edges, deg, 2 * nE);
    count_scatter_kernel<<<grid1(3 * nT), BLK, 0, stream>>>(tri_edges, tdeg, 3 * nT);
    clamp1_kernel<<<grid1(nN), BLK, 0, stream>>>(deg, nN);
    clamp1_kernel<<<grid1(nE), BLK, 0, stream>>>(tdeg, nE);

    // --- evolving feature copies ---
    hipMemcpyAsync(ef, ef_in, nE * 4 * 4, hipMemcpyDeviceToDevice, stream);
    hipMemcpyAsync(tf, tf_in, nT * 4 * 4, hipMemcpyDeviceToDevice, stream);

    // --- encoder: nf = tanh(tanh(in@W1+b1)@W2+b2) ---
    enc1_kernel<<<grid1(nN * 64), BLK, 0, stream>>>(uv, curv, enc_W1, enc_b1, hn, nN);
    gemm(hn, Penc2, enc_b2, 64, nf, 64, 64, nfb, 64, 64, nN, 64, 4, /*tanh*/1);

    // --- layers ---
    for (int l = 0; l < NL; ++l) {
        hipMemsetAsync(ns,   0, nN * 64 * 4, stream);
        hipMemsetAsync(es,   0, nE * 4 * 4, stream);
        hipMemsetAsync(ts,   0, nT * 4 * 4, stream);
        hipMemsetAsync(nfe,  0, nN * 64 * 4, stream);
        hipMemsetAsync(eacc, 0, nE * 64 * 4, stream);

        // Laplacian smoothing
        spmm64_kernel<<<grid1(NNZ0 * 4), BLK, 0, stream>>>(l0r, l0c, l0v, nf, ns, NNZ0);
        spmm4_kernel<<<grid1(NNZ1), BLK, 0, stream>>>(l1r, l1c, l1v, ef, es, NNZ1);
        spmm4_kernel<<<grid1(NNZ2), BLK, 0, stream>>>(l2r, l2c, l2v, tf, ts, NNZ2);

        // messages
        e2n_scatter_kernel<<<grid1(nE * 4), BLK, 0, stream>>>(
            ef, edges, e2n_W + (long)l * 256, e2n_b + (long)l * 64, nfe, nE);
        gemm(nfb, Pn2e[l], n2e_b + (long)l * 64, 64,
             nmsg, 64, 64, nullptr, 0, 0, nN, 64, 4, 0);
        t2e_scatter_kernel<<<grid1(nT * 4), BLK, 0, stream>>>(
            tf, tri_edges, t2e_W + (long)l * 256, t2e_b + (long)l * 64, eacc, nT);
        edge_combine_kernel<<<grid1(nE * 4), BLK, 0, stream>>>(
            eacc, tdeg, nmsg, edges, ef, Xe, nE);
        xn_build_kernel<<<grid1(nN * 64), BLK, 0, stream>>>(nfb, nfe, deg, Xn, nN);
        tri_build_kernel<<<grid1(nT * 4), BLK, 0, stream>>>(
            ef, tf, tri_edges, e2t_W + (long)l * 256, e2t_b + (long)l * 64, Xt, nT);

        // update MLPs (WMMA)
        gemm(Xn, Pnu1[l], nu_b1 + (long)l * 64, 64,
             nullptr, 0, 0, hn, 64, 64, nN, 128, 4, 1);
        gemm(hn, Pnu2[l], nu_b2 + (long)l * 64, 64,
             nmsg, 64, 64, nullptr, 0, 0, nN, 64, 4, 0);      // nmsg reused as node upd
        gemm(Xe, Peu1[l], eu_b1 + (long)l * 64, 64,
             nullptr, 0, 0, he, 64, 64, nE, 96, 4, 1);
        gemm(he, Peu2[l], eu_b2 + (long)l * 4, 4,
             eacc, 16, 4, nullptr, 0, 0, nE, 64, 1, 0);       // eacc reused as edge upd
        gemm(Xt, Ptu1[l], tu_b1 + (long)l * 64, 64,
             nullptr, 0, 0, ht, 64, 64, nT, 96, 4, 1);
        gemm(ht, Ptu2[l], tu_b2 + (long)l * 4, 4,
             (float*)Xt, 16, 4, nullptr, 0, 0, nT, 64, 1, 0); // Xt reused as tri upd

        // residual updates
        node_update_kernel<<<grid1(nN * 64), BLK, 0, stream>>>(nf, nfb, nmsg, ns, nN * 64);
        feat4_update_kernel<<<grid1(nE * 4), BLK, 0, stream>>>(ef, eacc, es, nE);
        feat4_update_kernel<<<grid1(nT * 4), BLK, 0, stream>>>(tf, (float*)Xt, ts, nT);
    }

    // --- decoder + ODE RHS ---
    gemm(nfb, Pdec1, dec_b1, 64, nullptr, 0, 0, hn, 64, 64, nN, 64, 4, 1);
    final_kernel<<<grid1(nN), BLK, 0, stream>>>(hn, dec_W2, dec_b2, uv, (float*)d_out, nN);
}
